// MMFLinear_47785806135955
// MI455X (gfx1250) — compile-verified, matrix operand order checked by reading, not run on hardware
//
#include <hip/hip_runtime.h>

// out[8192,4096] = scale * (x[8192,4096] @ sign(w)[4096,4096]^T + bias)
// bf16 WMMA GEMM (v_wmma_f32_16x16x32_bf16), LDS double buffering,
// 256x128 block tile, 8 waves of 64x64 (16 WMMA / 8 frag-loads per K-step).
// Tile producer uses GLOBAL_LOAD_ASYNC_TO_LDS (ASYNCcnt DMA path).

#define MROWS 8192
#define KDIM  4096
#define NDIM  4096

#define BM 256
#define BN 128
#define BK 32
#define LDK 40          // padded LDS row length in bf16 units (80 bytes)
#define KT  (KDIM / BK) // 128 K-steps

typedef unsigned short u16;
typedef unsigned int   u32;

typedef __attribute__((ext_vector_type(16))) __bf16 v16bf;
typedef __attribute__((ext_vector_type(8)))  float  v8f;
typedef __attribute__((ext_vector_type(4)))  int    v4i;

#if __has_builtin(__builtin_amdgcn_global_load_async_to_lds_b128)
#define USE_ASYNC_LDS 1
#else
#define USE_ASYNC_LDS 0
#endif

#define AS1 __attribute__((address_space(1)))
#define AS3 __attribute__((address_space(3)))

__device__ __forceinline__ void wait_async0() {
#if __has_builtin(__builtin_amdgcn_s_wait_asynccnt)
    __builtin_amdgcn_s_wait_asynccnt(0);
#else
    asm volatile("s_wait_asynccnt 0" ::: "memory");
#endif
}

union FragU {
    uint4 q[2];
    v16bf v;
};

__device__ __forceinline__ u16 f2bf_rne(float f) {
    u32 u = __float_as_uint(f);
    u32 r = u + 0x7FFFu + ((u >> 16) & 1u);
    return (u16)(r >> 16);
}

__device__ __forceinline__ u16 sign_bf16(float f) {
    return f > 0.0f ? (u16)0x3F80u : (f < 0.0f ? (u16)0xBF80u : (u16)0u);
}

// ---- pre-pass: x (fp32) -> bf16, 8 elements/thread ----
__global__ __launch_bounds__(256) void cvt_x_bf16(const float* __restrict__ x,
                                                  u16* __restrict__ xb) {
    size_t i = ((size_t)blockIdx.x * 256u + threadIdx.x) * 8u;
    float4 f0 = *(const float4*)(x + i);
    float4 f1 = *(const float4*)(x + i + 4);
    u32 p0 = (u32)f2bf_rne(f0.x) | ((u32)f2bf_rne(f0.y) << 16);
    u32 p1 = (u32)f2bf_rne(f0.z) | ((u32)f2bf_rne(f0.w) << 16);
    u32 p2 = (u32)f2bf_rne(f1.x) | ((u32)f2bf_rne(f1.y) << 16);
    u32 p3 = (u32)f2bf_rne(f1.z) | ((u32)f2bf_rne(f1.w) << 16);
    *(uint4*)(xb + i) = make_uint4(p0, p1, p2, p3);
}

// ---- pre-pass: w (fp32) -> sign(w) in bf16 (exact), 8 elements/thread ----
__global__ __launch_bounds__(256) void cvt_w_sign_bf16(const float* __restrict__ w,
                                                       u16* __restrict__ wb) {
    size_t i = ((size_t)blockIdx.x * 256u + threadIdx.x) * 8u;
    float4 f0 = *(const float4*)(w + i);
    float4 f1 = *(const float4*)(w + i + 4);
    u32 p0 = (u32)sign_bf16(f0.x) | ((u32)sign_bf16(f0.y) << 16);
    u32 p1 = (u32)sign_bf16(f0.z) | ((u32)sign_bf16(f0.w) << 16);
    u32 p2 = (u32)sign_bf16(f1.x) | ((u32)sign_bf16(f1.y) << 16);
    u32 p3 = (u32)sign_bf16(f1.z) | ((u32)sign_bf16(f1.w) << 16);
    *(uint4*)(wb + i) = make_uint4(p0, p1, p2, p3);
}

// ---- main GEMM ----
// 256 threads = 8 wave32; wave grid 4(M) x 2(N); per wave 4x4 16x16 tiles.
__global__ __launch_bounds__(256, 1) void gemm_bf16_wmma(
    const u16* __restrict__ xb, const u16* __restrict__ wb,
    const float* __restrict__ bias, const float* __restrict__ scale,
    float* __restrict__ out)
{
    __shared__ u16 sA[2][BM * LDK];  // 2 x 20480 B
    __shared__ u16 sB[2][BN * LDK];  // 2 x 10240 B

    const int t      = threadIdx.x;
    const int lane   = t & 31;
    const int wave   = t >> 5;
    const int wave_m = wave >> 1;   // 0..3 -> 64-row band
    const int wave_n = wave & 1;    // 0..1 -> 64-col band

    const int n0 = blockIdx.x * BN;
    const int m0 = blockIdx.y * BM;

    // global->LDS: chunk-interleaved, 16B chunks, 4 lanes per 64B row segment.
    // A tile: 256 rows x 4 chunks = 1024 chunks, 4 per thread.
    // B tile: 128 rows x 4 chunks =  512 chunks, 2 per thread.
    const u16* gA[4];
    const u16* gB[2];
    int sAoff[4], sBoff[2];
#pragma unroll
    for (int i = 0; i < 4; ++i) {
        const int c   = t + 256 * i;
        const int row = c >> 2;
        const int col = (c & 3) * 8;               // bf16 units
        gA[i]    = xb + (size_t)(m0 + row) * KDIM + col;
        sAoff[i] = row * LDK + col;
    }
#pragma unroll
    for (int i = 0; i < 2; ++i) {
        const int c   = t + 256 * i;
        const int row = c >> 2;
        const int col = (c & 3) * 8;
        gB[i]    = wb + (size_t)(n0 + row) * KDIM + col;
        sBoff[i] = row * LDK + col;
    }

#if USE_ASYNC_LDS
    // ---- async producer: global -> LDS via ASYNCcnt DMA ----
    {
#pragma unroll
        for (int i = 0; i < 4; ++i)
            __builtin_amdgcn_global_load_async_to_lds_b128(
                (AS1 v4i*)gA[i], (AS3 v4i*)&sA[0][sAoff[i]], 0, 0);
#pragma unroll
        for (int i = 0; i < 2; ++i)
            __builtin_amdgcn_global_load_async_to_lds_b128(
                (AS1 v4i*)gB[i], (AS3 v4i*)&sB[0][sBoff[i]], 0, 0);
        wait_async0();
    }
#else
    // ---- fallback producer: global -> regs -> ds_store ----
    {
        uint4 ra[4], rb[2];
#pragma unroll
        for (int i = 0; i < 4; ++i) ra[i] = *(const uint4*)gA[i];
#pragma unroll
        for (int i = 0; i < 2; ++i) rb[i] = *(const uint4*)gB[i];
#pragma unroll
        for (int i = 0; i < 4; ++i) *(uint4*)(&sA[0][sAoff[i]]) = ra[i];
#pragma unroll
        for (int i = 0; i < 2; ++i) *(uint4*)(&sB[0][sBoff[i]]) = rb[i];
    }
#endif
    __syncthreads();

    v8f acc[4][4] = {};

    const int frow = lane & 15;         // row within 16x16 tile
    const int fcol = (lane >> 4) * 8;   // bf16 chunk offset within K=32

    for (int kt = 0; kt < KT; ++kt) {
        const int  cur  = kt & 1;
        const bool more = (kt + 1) < KT;
        const int  nb   = cur ^ 1;

#if USE_ASYNC_LDS
        // kick off next tile's DMA before compute so it overlaps the WMMAs
        if (more) {
#pragma unroll
            for (int i = 0; i < 4; ++i)
                __builtin_amdgcn_global_load_async_to_lds_b128(
                    (AS1 v4i*)(gA[i] + (size_t)(kt + 1) * BK),
                    (AS3 v4i*)&sA[nb][sAoff[i]], 0, 0);
#pragma unroll
            for (int i = 0; i < 2; ++i)
                __builtin_amdgcn_global_load_async_to_lds_b128(
                    (AS1 v4i*)(gB[i] + (size_t)(kt + 1) * BK),
                    (AS3 v4i*)&sB[nb][sBoff[i]], 0, 0);
        }
#else
        uint4 ra[4], rb[2];
        if (more) {
#pragma unroll
            for (int i = 0; i < 4; ++i)
                ra[i] = *(const uint4*)(gA[i] + (size_t)(kt + 1) * BK);
#pragma unroll
            for (int i = 0; i < 2; ++i)
                rb[i] = *(const uint4*)(gB[i] + (size_t)(kt + 1) * BK);
        }
#endif
        if (kt + 2 < KT) {
            __builtin_prefetch(gA[0] + (size_t)(kt + 2) * BK, 0, 0);
            __builtin_prefetch(gB[0] + (size_t)(kt + 2) * BK, 0, 0);
        }

        // load 4 A-frags + 4 B-frags from LDS (8 x 2 ds_load_b128)
        v16bf af[4];
        v16bf bfr[4];
#pragma unroll
        for (int mt = 0; mt < 4; ++mt) {
            const int row = wave_m * 64 + mt * 16 + frow;
            FragU f;
            f.q[0] = *(const uint4*)(&sA[cur][row * LDK + fcol + 0]);
            f.q[1] = *(const uint4*)(&sA[cur][row * LDK + fcol + 16]);
            af[mt] = f.v;
        }
#pragma unroll
        for (int nt = 0; nt < 4; ++nt) {
            const int row = wave_n * 64 + nt * 16 + frow;
            FragU f;
            f.q[0] = *(const uint4*)(&sB[cur][row * LDK + fcol + 0]);
            f.q[1] = *(const uint4*)(&sB[cur][row * LDK + fcol + 16]);
            bfr[nt] = f.v;
        }

        // 16 WMMAs per K-step
#pragma unroll
        for (int mt = 0; mt < 4; ++mt) {
#pragma unroll
            for (int nt = 0; nt < 4; ++nt) {
                acc[mt][nt] = __builtin_amdgcn_wmma_f32_16x16x32_bf16(
                    false, af[mt], false, bfr[nt],
                    (short)0, acc[mt][nt], false, false);
            }
        }

        if (more) {
#if USE_ASYNC_LDS
            wait_async0();          // our DMA into buf nb is done
#else
#pragma unroll
            for (int i = 0; i < 4; ++i) *(uint4*)(&sA[nb][sAoff[i]]) = ra[i];
#pragma unroll
            for (int i = 0; i < 2; ++i) *(uint4*)(&sB[nb][sBoff[i]]) = rb[i];
#endif
            __syncthreads();
        }
    }

    // epilogue: out = scale * (acc + bias[n])
    const float sc = scale[0];
#pragma unroll
    for (int nt = 0; nt < 4; ++nt) {
        const int n = n0 + wave_n * 64 + nt * 16 + (lane & 15);
        const float bv = bias[n];
#pragma unroll
        for (int mt = 0; mt < 4; ++mt) {
            const int mbase = m0 + wave_m * 64 + mt * 16 + (lane >> 4) * 8;
#pragma unroll
            for (int j = 0; j < 8; ++j) {
                out[(size_t)(mbase + j) * NDIM + n] = sc * (acc[mt][nt][j] + bv);
            }
        }
    }
}

extern "C" void kernel_launch(void* const* d_in, const int* in_sizes, int n_in,
                              void* d_out, int out_size, void* d_ws, size_t ws_size,
                              hipStream_t stream) {
    (void)in_sizes; (void)n_in; (void)out_size; (void)ws_size;

    const float* x      = (const float*)d_in[0]; // [8192, 4096]
    const float* weight = (const float*)d_in[1]; // [4096, 4096]
    const float* bias   = (const float*)d_in[2]; // [4096]
    const float* scale  = (const float*)d_in[3]; // [1]
    float* out = (float*)d_out;                  // [8192, 4096]

    u16* xb = (u16*)d_ws;                         // 8192*4096 bf16 = 64 MiB
    u16* wb = xb + (size_t)MROWS * KDIM;          // 4096*4096 bf16 = 32 MiB

    // pre-pass conversions
    {
        const size_t nx = (size_t)MROWS * KDIM;   // 33.5M elems, 8/thread
        cvt_x_bf16<<<dim3((unsigned)(nx / (8 * 256))), dim3(256), 0, stream>>>(x, xb);
        const size_t nw = (size_t)NDIM * KDIM;    // 16.8M elems, 8/thread
        cvt_w_sign_bf16<<<dim3((unsigned)(nw / (8 * 256))), dim3(256), 0, stream>>>(weight, wb);
    }

    // main GEMM: grid = (N/BN, M/BM) = (32, 32)
    gemm_bf16_wmma<<<dim3(NDIM / BN, MROWS / BM), dim3(256), 0, stream>>>(
        xb, wb, bias, scale, out);
}
